// CombinactNN_19155554140562
// MI455X (gfx1250) — matrix operand order repaired and checked by reading, not compiled, not run on hardware
//
#include <hip/hip_runtime.h>
#include <hip/hip_bf16.h>

// ---------------- problem constants (from reference) ----------------
#define L_   3
#define M_   4096
#define K_   2
#define P_   4
#define G_   4
#define NUM_IN_  8192
#define NUM_OUT_ 1000
#define B_   2048
#define F_   6
#define EPS_BN_ 1e-5f
#define EPS_    1e-12f

#define CIN_G_  (NUM_IN_ / G_)        // 2048
#define MG_     (M_ / G_)             // 1024
#define D_LAST_ (M_ / K_ * P_)        // 8192
#define NOUT_PAD_ 1024                // padded rows for packed W_out

// ---------------- vector types for WMMA ----------------
typedef __attribute__((ext_vector_type(8)))  float   v8f;
typedef __attribute__((ext_vector_type(8)))  __bf16  v8bf;
typedef __attribute__((ext_vector_type(16))) __bf16  v16bf;

// ---------------- bf16 helpers (bit-exact RNE, no __bf16 arithmetic) ---------
static __device__ __forceinline__ unsigned short f32_to_bf16_bits(float f) {
    unsigned int u = __float_as_uint(f);
    unsigned int r = (u + 0x7FFFu + ((u >> 16) & 1u)) >> 16;
    return (unsigned short)r;
}

// ---------------- plain convert f32 -> bf16 (vectorized, grid-stride) --------
__global__ void cvt_f32_bf16_kernel(const float* __restrict__ src,
                                    unsigned short* __restrict__ dst,
                                    long n4) {
    long i = (long)blockIdx.x * blockDim.x + threadIdx.x;
    long stride = (long)gridDim.x * blockDim.x;
    for (; i < n4; i += stride) {
        float4 f = ((const float4*)src)[i];
        unsigned int a = f32_to_bf16_bits(f.x);
        unsigned int b = f32_to_bf16_bits(f.y);
        unsigned int c = f32_to_bf16_bits(f.z);
        unsigned int d = f32_to_bf16_bits(f.w);
        uint2 o;
        o.x = a | (b << 16);
        o.y = c | (d << 16);
        ((uint2*)dst)[i] = o;
    }
}

// ------- pack-convert: dst[row][g*(rowlen/4)+c] = bf16(src[row][c*4+g]) ------
__global__ void pack_cvt_kernel(const float* __restrict__ src,
                                unsigned short* __restrict__ dst,
                                int rows, int rowlen) {
    const int qc = rowlen >> 2;            // columns per group
    long total = (long)rows * qc;
    long i = (long)blockIdx.x * blockDim.x + threadIdx.x;
    long stride = (long)gridDim.x * blockDim.x;
    for (; i < total; i += stride) {
        int row = (int)(i / qc);
        int c   = (int)(i - (long)row * qc);
        float4 f = ((const float4*)src)[i];  // src[row][4c .. 4c+3]
        unsigned short* drow = dst + (size_t)row * rowlen;
        drow[0 * qc + c] = f32_to_bf16_bits(f.x);
        drow[1 * qc + c] = f32_to_bf16_bits(f.y);
        drow[2 * qc + c] = f32_to_bf16_bits(f.z);
        drow[3 * qc + c] = f32_to_bf16_bits(f.w);
    }
}

// ---------------- LDS fragment loaders (ISA 7.12.2 layouts) ----------------
static __device__ __forceinline__ v16bf load_a_frag(const unsigned short* rowp, int half) {
    union { uint4 u; v8bf v; } c1, c2;
    c1.u = *(const uint4*)(rowp + half * 8);
    c2.u = *(const uint4*)(rowp + 16 + half * 8);
    v16bf f;
#pragma unroll
    for (int i = 0; i < 8; ++i) { f[i] = c1.v[i]; f[i + 8] = c2.v[i]; }
    return f;
}
static __device__ __forceinline__ v16bf load_b_frag(const unsigned short* rowp, int half) {
    union { uint4 u; v8bf v; } c1, c2;
    c1.u = *(const uint4*)(rowp + half * 8);
    c2.u = *(const uint4*)(rowp + 16 + half * 8);
    v16bf f;
#pragma unroll
    for (int i = 0; i < 8; ++i) { f[i] = c1.v[i]; f[i + 8] = c2.v[i]; }
    return f;
}

// ---------------- GEMM:  C = A(BxK) * Bw(NxK)^T + bias ----------------------
// Block tile 128x64x32, 128 threads = 4 waves; wave owns 32 rows (2 A-frags),
// reuses each B-frag twice -> 8 WMMAs per K-step per wave. Double-buffered
// LDS staged via gfx1250 async global->LDS b128 copies (ASYNCcnt), next-tile
// loads overlapped with current-tile WMMAs.
#define BM 128
#define BN 64
#define BK 32
#define LDS_PITCH 48   // bf16 elems per row: 96B row stride, 16B-aligned chunks

static __device__ __forceinline__ void async_stage_tile(
    const unsigned short* __restrict__ gA,   // -> (b0, k0) incl. group col off
    int ldA,
    const unsigned short* __restrict__ gB,   // -> (n0, k0)
    int ldB,
    unsigned short* sAbuf, unsigned short* sBbuf, int t) {
    // A tile: 128 rows x 32 cols = 512 16B-chunks -> 4 per thread
#pragma unroll
    for (int i = 0; i < 4; ++i) {
        int q = t + i * 128;                 // 0..511
        int row = q >> 2, colc = (q & 3) * 8;
        unsigned ldsA = (unsigned)(uintptr_t)(sAbuf + row * LDS_PITCH + colc);
        unsigned long long ga = (unsigned long long)(uintptr_t)(gA + (size_t)row * ldA + colc);
        asm volatile("global_load_async_to_lds_b128 %0, %1, off"
                     :: "v"(ldsA), "v"(ga) : "memory");
    }
    // B tile: 64 rows x 32 cols = 256 16B-chunks -> 2 per thread
#pragma unroll
    for (int i = 0; i < 2; ++i) {
        int q = t + i * 128;                 // 0..255
        int row = q >> 2, colc = (q & 3) * 8;
        unsigned ldsB = (unsigned)(uintptr_t)(sBbuf + row * LDS_PITCH + colc);
        unsigned long long gb = (unsigned long long)(uintptr_t)(gB + (size_t)row * ldB + colc);
        asm volatile("global_load_async_to_lds_b128 %0, %1, off"
                     :: "v"(ldsB), "v"(gb) : "memory");
    }
}

__global__ __launch_bounds__(128)
void gemm_bf16_nt(const unsigned short* __restrict__ A, int ldA, int aGroupColOff,
                  const unsigned short* __restrict__ Bw, size_t bwGroupStride,
                  const float* __restrict__ bias, int biasGroupStride,
                  float* __restrict__ C, int ldC, size_t cGroupStride,
                  int N, int Kdim) {
    __shared__ __attribute__((aligned(16))) unsigned short sA[2][BM * LDS_PITCH];
    __shared__ __attribute__((aligned(16))) unsigned short sB[2][BN * LDS_PITCH];

    const int g  = blockIdx.z;
    const int n0 = blockIdx.x * BN;
    const int b0 = blockIdx.y * BM;
    const unsigned short* Ap = A + (size_t)b0 * ldA + (size_t)g * aGroupColOff;
    const unsigned short* Bp = Bw + (size_t)g * bwGroupStride + (size_t)n0 * Kdim;
    const float* biasp = bias + (size_t)g * biasGroupStride;
    float* Cp = C + (size_t)g * cGroupStride;

    const int t    = threadIdx.x;
    const int wv   = t >> 5;
    const int lane = t & 31;
    const int half = lane >> 4;
    const int r    = lane & 15;

    v8f zero8 = {0.f, 0.f, 0.f, 0.f, 0.f, 0.f, 0.f, 0.f};
    v8f acc[2][4];
#pragma unroll
    for (int mt = 0; mt < 2; ++mt)
#pragma unroll
        for (int nt = 0; nt < 4; ++nt) acc[mt][nt] = zero8;

    const int nIter = Kdim / BK;
    async_stage_tile(Ap, ldA, Bp, Kdim, sA[0], sB[0], t);

    for (int it = 0; it < nIter; ++it) {
        const int cur = it & 1;
        const bool hasNext = (it + 1) < nIter;
        if (hasNext) {
            int k0n = (it + 1) * BK;
            async_stage_tile(Ap + k0n, ldA, Bp + k0n, Kdim,
                             sA[cur ^ 1], sB[cur ^ 1], t);
            // retire current tile's 6 per-thread async loads; next 6 stay in flight
            asm volatile("s_wait_asynccnt 6" ::: "memory");
        } else {
            asm volatile("s_wait_asynccnt 0" ::: "memory");
        }
        __syncthreads();

        v16bf af0 = load_a_frag(&sA[cur][(wv * 32 + r) * LDS_PITCH], half);
        v16bf af1 = load_a_frag(&sA[cur][(wv * 32 + 16 + r) * LDS_PITCH], half);
#pragma unroll
        for (int nt = 0; nt < 4; ++nt) {
            v16bf bfrag = load_b_frag(&sB[cur][(nt * 16 + r) * LDS_PITCH], half);
            acc[0][nt] = __builtin_amdgcn_wmma_f32_16x16x32_bf16(
                false, af0, false, bfrag, (short)0, acc[0][nt], false, false);
            acc[1][nt] = __builtin_amdgcn_wmma_f32_16x16x32_bf16(
                false, af1, false, bfrag, (short)0, acc[1][nt], false, false);
        }
        __syncthreads();   // protect buffer 'cur' from being restaged early
    }

    // --- epilogue: D layout -> m_local = v + 8*half, n_local = r ---
#pragma unroll
    for (int mt = 0; mt < 2; ++mt) {
#pragma unroll
        for (int nt = 0; nt < 4; ++nt) {
            int n = n0 + nt * 16 + r;
            if (n < N) {
                float bv = biasp[n];
#pragma unroll
                for (int v = 0; v < 8; ++v) {
                    int brow = b0 + wv * 32 + mt * 16 + v + 8 * half;
                    Cp[(size_t)brow * ldC + n] = acc[mt][nt][v] + bv;
                }
            }
        }
    }
}

// ---------------- batch-norm statistics over B (per channel) ----------------
__global__ __launch_bounds__(512)
void bn_stats_kernel(const float* __restrict__ ypack,
                     float* __restrict__ mean, float* __restrict__ rstd) {
    const int g  = blockIdx.y;
    const int n0 = blockIdx.x * 64;
    const int t  = threadIdx.x;
    const int c  = t & 63;
    const int rr = t >> 6;                 // 0..7
    const float* base = ypack + (size_t)g * B_ * MG_ + n0 + c;
    float s = 0.f, s2 = 0.f;
    for (int b = rr; b < B_; b += 8) {
        float v = base[(size_t)b * MG_];
        s += v; s2 += v * v;
    }
    __shared__ float shS[8][64];
    __shared__ float shS2[8][64];
    shS[rr][c] = s; shS2[rr][c] = s2;
    __syncthreads();
    if (rr == 0) {
#pragma unroll
        for (int i = 1; i < 8; ++i) { s += shS[i][c]; s2 += shS2[i][c]; }
        float mu  = s / (float)B_;
        float var = s2 / (float)B_ - mu * mu;
        int j = (n0 + c) * G_ + g;         // channel index in (B, M) layout
        mean[j] = mu;
        rstd[j] = rsqrtf(var + EPS_BN_);
    }
}

// ------- normalize + permute + combinact -> next h (bf16, group-packed) -----
__global__ __launch_bounds__(256)
void act_kernel(const float* __restrict__ ypack,
                const float* __restrict__ mean, const float* __restrict__ rstd,
                const float* __restrict__ gamma, const float* __restrict__ beta,
                const int* __restrict__ perm,     // (P, M)
                const float* __restrict__ alpha,  // (P, F)
                unsigned short* __restrict__ hout /* (B, 8192) packed bf16 */) {
    int idx = blockIdx.x * blockDim.x + threadIdx.x;
    if (idx >= B_ * (M_ / K_) * P_) return;
    const int c = idx & (M_ / K_ - 1);        // 0..2047
    const int p = (idx >> 11) & (P_ - 1);     // 0..3
    const int b = idx >> 13;                  // 0..2047

    float z[K_];
#pragma unroll
    for (int k = 0; k < K_; ++k) {
        int j = perm[p * M_ + c * K_ + k];
        int gg = j & (G_ - 1);
        int nn = j >> 2;
        float v = ypack[(size_t)gg * B_ * MG_ + (size_t)b * MG_ + nn];
        z[k] = (v - mean[j]) * rstd[j] * gamma[j] + beta[j];
    }
    const float z0 = z[0], z1 = z[1];

    float a0 = fmaxf(z0, z1);                                   // max
    float pr = z0 * z1;
    float sg = (pr > 0.f) ? 1.f : ((pr < 0.f) ? -1.f : 0.f);
    float a1 = sg * sqrtf(fabsf(pr) + EPS_);                    // signed geo-mean
    float s0 = 1.f / (1.f + __expf(-z0));
    float s1 = 1.f / (1.f + __expf(-z1));
    float a2 = z0 * s0 * s1;                                    // swish-like
    float a3 = sqrtf(z0 * z0 + z1 * z1 + EPS_);                 // l2
    float a4 = fmaxf(fabsf(z0), fabsf(z1));                     // linf
    float mx = a0;
    float a5 = mx + __logf(__expf(z0 - mx) + __expf(z1 - mx));  // logsumexp

    float al[F_];
    float am = -1e30f;
#pragma unroll
    for (int f = 0; f < F_; ++f) { al[f] = alpha[p * F_ + f]; am = fmaxf(am, al[f]); }
    float den = 0.f;
#pragma unroll
    for (int f = 0; f < F_; ++f) { al[f] = __expf(al[f] - am); den += al[f]; }
    float inv = 1.f / den;
    float outv = (a0 * al[0] + a1 * al[1] + a2 * al[2] +
                  a3 * al[3] + a4 * al[4] + a5 * al[5]) * inv;

    // packed write: j = p*2048 + c  ->  group g = c&3, packed col = p*512 + c/4
    hout[(size_t)b * D_LAST_ + (size_t)(c & 3) * (D_LAST_ / G_) +
         p * (D_LAST_ / G_ / P_) + (c >> 2)] = f32_to_bf16_bits(outv);
}

// ---------------- host-side orchestration ----------------
static inline size_t align256(size_t x) { return (x + 255) & ~(size_t)255; }

extern "C" void kernel_launch(void* const* d_in, const int* in_sizes, int n_in,
                              void* d_out, int out_size, void* d_ws, size_t ws_size,
                              hipStream_t stream) {
    const float* x     = (const float*)d_in[0];   // (B, 8192)
    const int*   perms = (const int*)  d_in[1];   // (L, P, M)
    const float* W     = (const float*)d_in[2];   // (L, G, MG, CIN_G)
    const float* bvec  = (const float*)d_in[3];   // (L, G, MG)
    const float* gamma = (const float*)d_in[4];   // (L, M)
    const float* beta  = (const float*)d_in[5];   // (L, M)
    const float* alpha = (const float*)d_in[6];   // (L, P, F)
    const float* Wout  = (const float*)d_in[7];   // (NUM_OUT, D_LAST)
    const float* bout  = (const float*)d_in[8];   // (NUM_OUT)
    float* out = (float*)d_out;                    // (B, NUM_OUT)

    // workspace carve-up
    char* p = (char*)d_ws;
    unsigned short* h0 = (unsigned short*)p;  p += align256((size_t)B_ * D_LAST_ * 2);
    unsigned short* h1 = (unsigned short*)p;  p += align256((size_t)B_ * D_LAST_ * 2);
    float* ypack       = (float*)p;           p += align256((size_t)B_ * M_ * 4);
    unsigned short* Wbf = (unsigned short*)p; p += align256((size_t)L_ * G_ * MG_ * CIN_G_ * 2);
    unsigned short* Woutp = (unsigned short*)p; p += align256((size_t)NOUT_PAD_ * D_LAST_ * 2);
    float* meanv = (float*)p;                 p += align256((size_t)M_ * 4);
    float* rstdv = (float*)p;                 p += align256((size_t)M_ * 4);
    (void)ws_size; (void)n_in; (void)in_sizes; (void)out_size;

    // 1) conversions: x and W_out get the group-packed column permutation,
    //    per-group W is already (N x K) contiguous.
    pack_cvt_kernel<<<4096, 256, 0, stream>>>(x, h0, B_, NUM_IN_);
    cvt_f32_bf16_kernel<<<4096, 256, 0, stream>>>(
        W, Wbf, (long)L_ * G_ * MG_ * CIN_G_ / 4);
    pack_cvt_kernel<<<4096, 256, 0, stream>>>(Wout, Woutp, NUM_OUT_, D_LAST_);

    // 2) layers
    unsigned short* hin = h0;
    unsigned short* hout = h1;
    for (int l = 0; l < L_; ++l) {
        dim3 grid(MG_ / BN, B_ / BM, G_);   // 16 x 16 x 4
        gemm_bf16_nt<<<grid, 128, 0, stream>>>(
            hin, NUM_IN_, /*aGroupColOff=*/CIN_G_,
            Wbf + (size_t)l * G_ * MG_ * CIN_G_, (size_t)MG_ * CIN_G_,
            bvec + (size_t)l * G_ * MG_, MG_,
            ypack, MG_, (size_t)B_ * MG_,
            /*N=*/MG_, /*K=*/CIN_G_);

        bn_stats_kernel<<<dim3(MG_ / 64, G_), 512, 0, stream>>>(ypack, meanv, rstdv);

        int total = B_ * (M_ / K_) * P_;
        act_kernel<<<(total + 255) / 256, 256, 0, stream>>>(
            ypack, meanv, rstdv,
            gamma + (size_t)l * M_, beta + (size_t)l * M_,
            perms + (size_t)l * P_ * M_, alpha + (size_t)l * P_ * F_,
            hout);

        unsigned short* tmp = hin; hin = hout; hout = tmp;
    }

    // 3) output GEMM: hp (B,8192) x Woutp(1000,8192)^T + b_out
    {
        dim3 grid((NOUT_PAD_) / BN, B_ / BM, 1);  // 16 x 16 x 1
        gemm_bf16_nt<<<grid, 128, 0, stream>>>(
            hin, D_LAST_, /*aGroupColOff=*/0,
            Woutp, 0,
            bout, 0,
            out, NUM_OUT_, 0,
            /*N=*/NUM_OUT_, /*K=*/D_LAST_);
    }
}
